// TransformerDecoder_18829136626007
// MI455X (gfx1250) — matrix operand order repaired and checked by reading, not compiled, not run on hardware
//
#include <hip/hip_runtime.h>
#include <hip/hip_bf16.h>
#include <math.h>
#include <stdint.h>

#define LNUM 6
#define BB   16
#define HH   16
#define DD   1024
#define DHD  64
#define FF   4096
#define SS   256
#define PP   255

#define KSTEP    64          // K rows DMA'd per LDS tile
#define TILECOLS 64          // N columns per block tile
#define LDSROW   65          // padded dwords per tile row (TDM pad: +1 dword / 64)
#define SPLITK   8           // K-chunks per GEMM (independent blocks -> HBM-wide streaming)

typedef __bf16          v16bf __attribute__((ext_vector_type(16)));
typedef __bf16          v8bf  __attribute__((ext_vector_type(8)));
typedef float           v8f   __attribute__((ext_vector_type(8)));
typedef unsigned int    v4u32 __attribute__((ext_vector_type(4)));
typedef int             v8i32 __attribute__((ext_vector_type(8)));
typedef int             v4i32 __attribute__((ext_vector_type(4)));

union ABf  { v16bf v; __bf16 b[16]; };
union AccU { v8f   v; float  f[8];  };

// --------------------------------------------------------------------------
// TDM: DMA a KSTEP x TILECOLS fp32 tile (row stride N floats in global) into
// LDS at ldsOff, inserting 1 pad dword per 64-dword row (LDS row stride = 65
// dwords -> conflict-free column reads). D# layout per CDNA5 ISA.
// --------------------------------------------------------------------------
__device__ __forceinline__ void tdm_load_tile(const float* gsrc, unsigned ldsOff, int N)
{
  unsigned long long ga = (unsigned long long)(uintptr_t)gsrc;
  v4u32 g0;
  g0[0] = 1u;                                                // count=1, user mode
  g0[1] = ldsOff;                                            // lds_addr
  g0[2] = (unsigned)ga;                                      // global_addr[31:0]
  g0[3] = ((unsigned)(ga >> 32) & 0x01FFFFFFu) | (2u << 30); // addr[56:32] | type=2
  v8i32 g1;
  g1[0] = (int)((2u << 16)        // data_size = 4B
              | (1u << 20)        // pad_enable
              | (5u << 22));      // pad_interval: 2<<5 = 64 dwords (pad_amount=0 -> 1 dw)
  g1[1] = (int)(((unsigned)TILECOLS & 0xFFFFu) << 16);       // tensor_dim0 lo16
  g1[2] = (int)(((unsigned)KSTEP   & 0xFFFFu) << 16);        // dim0 hi=0 | tensor_dim1 lo16
  g1[3] = (int)(((unsigned)TILECOLS) << 16);                 // dim1 hi=0 | tile_dim0
  g1[4] = (int)((unsigned)KSTEP);                            // tile_dim1 | tile_dim2=0
  g1[5] = (int)N;                                            // tensor_dim0_stride lo32
  g1[6] = 0;
  g1[7] = 0;
  v4i32 z4 = {0, 0, 0, 0};
#if defined(__clang_major__) && (__clang_major__ >= 23)
  v8i32 z8 = {0, 0, 0, 0, 0, 0, 0, 0};
  __builtin_amdgcn_tensor_load_to_lds(g0, g1, z4, z4, z8, 0);
#else
  __builtin_amdgcn_tensor_load_to_lds(g0, g1, z4, z4, 0);
#endif
}

// --------------------------------------------------------------------------
// Split-K partial GEMM: part[z][split][16 x N] = scale_z * A[16 x Kchunk] @ Wz
// grid = (N/64, SPLITK, Z).  Z=3 fuses the q/k/v projections (shared A) into
// one launch so their weight streams overlap on the HBM side.
// Per block: double-buffered TDM DMA of 64x64 fp32 weight slabs into LDS,
// f32->bf16 convert in-register, v_wmma_f32_16x16x32_bf16 accumulate.
// --------------------------------------------------------------------------
__global__ void gemm_partial(const __bf16* __restrict__ A,
                             const float* __restrict__ W0,
                             const float* __restrict__ W1,
                             const float* __restrict__ W2,
                             float* __restrict__ part,
                             int K, int N, float s0, float s1, float s2)
{
  __shared__ float wtile[2][KSTEP * LDSROW];

  const int z     = blockIdx.z;
  const float* W  = (z == 0) ? W0 : ((z == 1) ? W1 : W2);
  const float sc  = (z == 0) ? s0 : ((z == 1) ? s1 : s2);
  const int split = blockIdx.y;
  const int kchunk = K / SPLITK;
  const int kbase  = split * kchunk;
  const int nIter  = kchunk / KSTEP;

  const int wave = threadIdx.x >> 5;
  const int lane = threadIdx.x & 31;
  const int n0b  = blockIdx.x * TILECOLS;
  const int row  = lane & 15;
  const int hi   = lane >> 4;
  const int col  = wave * 16 + row;     // column within 64-wide block tile
  const int ncol = n0b + col;

  AccU acc;
#pragma unroll
  for (int i = 0; i < 8; ++i) acc.f[i] = 0.0f;

  const __bf16* arow = A + (size_t)row * K + kbase;

  if (threadIdx.x < 32)
    tdm_load_tile(W + (size_t)kbase * N + n0b, (unsigned)(uintptr_t)&wtile[0][0], N);

  for (int it = 0; it < nIter; ++it) {
    if (threadIdx.x < 32) {
      if (it + 1 < nIter) {
        tdm_load_tile(W + (size_t)(kbase + (it + 1) * KSTEP) * N + n0b,
                      (unsigned)(uintptr_t)&wtile[(it + 1) & 1][0], N);
        __builtin_amdgcn_s_wait_tensorcnt((short)1);   // prev DMA complete
      } else {
        __builtin_amdgcn_s_wait_tensorcnt((short)0);
      }
    }
    __syncthreads();                                   // publish LDS tile

    const float* wt = &wtile[it & 1][0];
    const int    k0 = it * KSTEP;
#pragma unroll
    for (int s = 0; s < 2; ++s) {
      // A fragment: ISA 7.12.2 16-bit A 16x32 layout -> two contiguous 16B runs
      ABf a;
      const int kb = k0 + s * 32 + hi * 8;
      v8bf lo  = *(const v8bf*)(arow + kb);
      v8bf hi8 = *(const v8bf*)(arow + kb + 16);
#pragma unroll
      for (int e = 0; e < 8; ++e) { a.b[e] = lo[e]; a.b[e + 8] = hi8[e]; }

      // B fragment from LDS: col = lane&15, K = 8*(lane>>4) + e (e<8), 16 + 8*(lane>>4) + (e-8) (e>=8)
      ABf bm;
      const int krow0 = s * 32 + hi * 8;
#pragma unroll
      for (int e = 0; e < 16; ++e)
        bm.b[e] = (__bf16)wt[(krow0 + (e < 8 ? e : e + 8)) * LDSROW + col];

      acc.v = __builtin_amdgcn_wmma_f32_16x16x32_bf16(
          false, a.v, false, bm.v, (short)0, acc.v, false, false);
    }
    __syncthreads();                                   // tile consumed, safe to overwrite
  }

  float* prow = part + ((size_t)(z * SPLITK + split) * 16) * N;
  const int m0 = hi * 8;
#pragma unroll
  for (int r = 0; r < 8; ++r)
    prow[(size_t)(m0 + r) * N + ncol] = acc.f[r] * sc;
}

// --------------------------------------------------------------------------
// Deterministic split-K reductions (fixed summation order, no atomics).
// --------------------------------------------------------------------------
__global__ void reduce_qkv(const float* __restrict__ part,   // [3][SPLITK][16*DD]
                           float* __restrict__ qb,
                           float* __restrict__ kb,
                           float* __restrict__ vb)
{
  const int z = blockIdx.z;
  const size_t i = (size_t)blockIdx.x * 256 + threadIdx.x;   // over 16*DD
  const float* p = part + (size_t)z * SPLITK * 16 * DD + i;
  float s = 0.0f;
#pragma unroll
  for (int k = 0; k < SPLITK; ++k) s += p[(size_t)k * 16 * DD];
  float* out = (z == 0) ? qb : ((z == 1) ? kb : vb);
  out[i] = s;
}

template<bool RES, bool RELU, bool WF, bool WB, bool BIAS>
__global__ void reduceK(const float* __restrict__ part,      // [SPLITK][16*N]
                        const float* __restrict__ bias,
                        float* __restrict__ outF,
                        __bf16* __restrict__ outB,
                        int N)
{
  const size_t i = (size_t)blockIdx.x * 256 + threadIdx.x;   // over 16*N
  const int n = (int)(i & (size_t)(N - 1));
  float s = 0.0f;
#pragma unroll
  for (int k = 0; k < SPLITK; ++k) s += part[(size_t)k * 16 * N + i];
  if (BIAS) s += bias[n];
  if (RELU) s = fmaxf(s, 0.0f);
  if (WF) { float o = s; if (RES) o += outF[i]; outF[i] = o; }
  if (WB) outB[i] = (__bf16)s;
}

// --------------------------------------------------------------------------
// One wave per (batch, head): softmax(q.K^T) @ V over 256 keys, DH=64.
// --------------------------------------------------------------------------
__global__ void attn16(const float* __restrict__ q,
                       const float* __restrict__ ck,   // [B,H,Pc,DH]
                       const float* __restrict__ cv,
                       const float* __restrict__ kx,   // [B, D] new key row (or null)
                       const float* __restrict__ vx,
                       const unsigned char* __restrict__ mask, // [B,S] or null
                       int Pc,
                       __bf16* __restrict__ out)
{
  __shared__ float sQ[DHD];
  __shared__ float sPart[32][DHD];
  const int lane = threadIdx.x;
  const int bh   = blockIdx.x;
  const int b    = bh >> 4;
  const int h    = bh & 15;

  const float* qrow = q + (size_t)b * DD + h * DHD;
  for (int i = lane; i < DHD; i += 32) sQ[i] = qrow[i];
  __syncthreads();

  float sc[8];
#pragma unroll
  for (int jj = 0; jj < 8; ++jj) {
    const int j = jj * 32 + lane;
    const float* krow = (j < Pc) ? (ck + ((size_t)bh * Pc + j) * DHD)
                                 : (kx + (size_t)b * DD + h * DHD);
    float d = 0.0f;
    for (int e = 0; e < DHD; ++e) d += sQ[e] * krow[e];
    if (mask && !mask[(size_t)b * SS + j]) d = -1.0e9f;
    sc[jj] = d;
  }

  float mx = sc[0];
#pragma unroll
  for (int jj = 1; jj < 8; ++jj) mx = fmaxf(mx, sc[jj]);
  for (int off = 16; off > 0; off >>= 1) mx = fmaxf(mx, __shfl_xor(mx, off, 32));

  float sum = 0.0f;
#pragma unroll
  for (int jj = 0; jj < 8; ++jj) { sc[jj] = __expf(sc[jj] - mx); sum += sc[jj]; }
  for (int off = 16; off > 0; off >>= 1) sum += __shfl_xor(sum, off, 32);
  const float inv = 1.0f / sum;

  float o[DHD];
#pragma unroll
  for (int e = 0; e < DHD; ++e) o[e] = 0.0f;
#pragma unroll
  for (int jj = 0; jj < 8; ++jj) {
    const int j = jj * 32 + lane;
    const float p = sc[jj] * inv;
    const float* vrow = (j < Pc) ? (cv + ((size_t)bh * Pc + j) * DHD)
                                 : (vx + (size_t)b * DD + h * DHD);
    for (int e = 0; e < DHD; ++e) o[e] += p * vrow[e];
  }
  for (int e = 0; e < DHD; ++e) sPart[lane][e] = o[e];
  __syncthreads();

  for (int c = lane; c < DHD; c += 32) {
    float s = 0.0f;
    for (int r = 0; r < 32; ++r) s += sPart[r][c];
    out[(size_t)b * DD + h * DHD + c] = (__bf16)s;
  }
}

// --------------------------------------------------------------------------
// LayerNorm of one 1024-wide row per block (16 blocks, 256 threads).
// --------------------------------------------------------------------------
__global__ void ln16(const float* __restrict__ x,
                     const float* __restrict__ g,
                     const float* __restrict__ bta,
                     __bf16* __restrict__ outB,
                     float* __restrict__ outF)
{
  __shared__ float red[256];
  const int t   = threadIdx.x;
  const int row = blockIdx.x;
  const float* xr = x + (size_t)row * DD;

  float v[4];
  float s = 0.0f;
#pragma unroll
  for (int i = 0; i < 4; ++i) { v[i] = xr[t + i * 256]; s += v[i]; }
  red[t] = s; __syncthreads();
  for (int o = 128; o > 0; o >>= 1) { if (t < o) red[t] += red[t + o]; __syncthreads(); }
  const float mu = red[0] * (1.0f / (float)DD);
  __syncthreads();

  float s2 = 0.0f;
#pragma unroll
  for (int i = 0; i < 4; ++i) { const float d = v[i] - mu; s2 += d * d; }
  red[t] = s2; __syncthreads();
  for (int o = 128; o > 0; o >>= 1) { if (t < o) red[t] += red[t + o]; __syncthreads(); }
  const float rinv = rsqrtf(red[0] * (1.0f / (float)DD) + 1e-6f);

#pragma unroll
  for (int i = 0; i < 4; ++i) {
    const int c = t + i * 256;
    const float y = (v[i] - mu) * rinv * g[c] + bta[c];
    if (outB) outB[(size_t)row * DD + c] = (__bf16)y;
    if (outF) outF[(size_t)row * DD + c] = y;
  }
}

__global__ void embed16(const int* __restrict__ tgt,
                        const float* __restrict__ emb,
                        const float* __restrict__ pe,
                        float* __restrict__ x)
{
  const int b = blockIdx.x;
  const int tok = tgt[b];
  for (int d = threadIdx.x; d < DD; d += 256)
    x[(size_t)b * DD + d] = emb[(size_t)tok * DD + d] + pe[d];
}

// --------------------------------------------------------------------------
extern "C" void kernel_launch(void* const* d_in, const int* in_sizes, int n_in,
                              void* d_out, int out_size, void* d_ws, size_t ws_size,
                              hipStream_t stream)
{
  (void)in_sizes; (void)n_in; (void)out_size; (void)ws_size;

  const int*   tgt   = (const int*)  d_in[0];
  const float* emb   = (const float*)d_in[1];
  const float* pe    = (const float*)d_in[2];
  const float* ln1g  = (const float*)d_in[3];
  const float* ln1b  = (const float*)d_in[4];
  const float* wqs   = (const float*)d_in[5];
  const float* wks   = (const float*)d_in[6];
  const float* wvs   = (const float*)d_in[7];
  const float* wos   = (const float*)d_in[8];
  const float* ln2g  = (const float*)d_in[9];
  const float* ln2b  = (const float*)d_in[10];
  const float* wqc   = (const float*)d_in[11];
  const float* woc   = (const float*)d_in[12];
  const float* flng  = (const float*)d_in[13];
  const float* flnb  = (const float*)d_in[14];
  const float* w1    = (const float*)d_in[15];
  const float* b1    = (const float*)d_in[16];
  const float* w2    = (const float*)d_in[17];
  const float* b2    = (const float*)d_in[18];
  const float* fg    = (const float*)d_in[19];
  const float* fb    = (const float*)d_in[20];
  const float* selfk = (const float*)d_in[21];
  const float* selfv = (const float*)d_in[22];
  const float* memk  = (const float*)d_in[23];
  const float* memv  = (const float*)d_in[24];
  const unsigned char* mask = (const unsigned char*)d_in[25];

  char* ws = (char*)d_ws;
  float*  x    = (float*)ws;   ws += (size_t)BB * DD * 4;
  __bf16* hbuf = (__bf16*)ws;  ws += (size_t)BB * DD * 2;
  float*  qb   = (float*)ws;   ws += (size_t)BB * DD * 4;
  float*  kb   = (float*)ws;   ws += (size_t)BB * DD * 4;
  float*  vb   = (float*)ws;   ws += (size_t)BB * DD * 4;
  __bf16* ab   = (__bf16*)ws;  ws += (size_t)BB * DD * 2;
  __bf16* fbuf = (__bf16*)ws;  ws += (size_t)BB * FF * 2;
  float*  part = (float*)ws;   ws += (size_t)SPLITK * BB * FF * 4;  // 2 MB, >= 3*SPLITK*16*DD

  const float SCALE = 0.125f;  // 1/sqrt(DH)

  dim3 tRow(256), gRow(BB);
  dim3 tG(128);
  dim3 gQKV(DD / 64, SPLITK, 3);     // fused q/k/v projection
  dim3 gG1 (DD / 64, SPLITK, 1);
  dim3 gGf (FF / 64, SPLITK, 1);
  dim3 gRqkv((BB * DD) / 256, 1, 3);
  dim3 gRd ((BB * DD) / 256);
  dim3 gRf ((BB * FF) / 256);
  dim3 gA(BB * HH), tA(32);

  embed16<<<gRow, tRow, 0, stream>>>(tgt, emb, pe, x);

  for (int l = 0; l < LNUM; ++l) {
    const size_t woff  = (size_t)l * DD * DD;
    const size_t skoff = (size_t)l * BB * HH * PP * DHD;
    const size_t mkoff = (size_t)l * BB * HH * SS * DHD;

    // ---- self attention ----
    ln16<<<gRow, tRow, 0, stream>>>(x, ln1g + (size_t)l * DD, ln1b + (size_t)l * DD, hbuf, nullptr);
    gemm_partial<<<gQKV, tG, 0, stream>>>(hbuf, wqs + woff, wks + woff, wvs + woff,
                                          part, DD, DD, SCALE, 1.0f, 1.0f);
    reduce_qkv<<<gRqkv, tRow, 0, stream>>>(part, qb, kb, vb);
    attn16<<<gA, tA, 0, stream>>>(qb, selfk + skoff, selfv + skoff, kb, vb, nullptr, PP, ab);
    gemm_partial<<<gG1, tG, 0, stream>>>(ab, wos + woff, wos + woff, wos + woff,
                                         part, DD, DD, 1.0f, 1.0f, 1.0f);
    reduceK<true, false, true, false, false><<<gRd, tRow, 0, stream>>>(part, nullptr, x, nullptr, DD);

    // ---- cross attention ----
    ln16<<<gRow, tRow, 0, stream>>>(x, ln2g + (size_t)l * DD, ln2b + (size_t)l * DD, hbuf, nullptr);
    gemm_partial<<<gG1, tG, 0, stream>>>(hbuf, wqc + woff, wqc + woff, wqc + woff,
                                         part, DD, DD, SCALE, 1.0f, 1.0f);
    reduceK<false, false, true, false, false><<<gRd, tRow, 0, stream>>>(part, nullptr, qb, nullptr, DD);
    attn16<<<gA, tA, 0, stream>>>(qb, memk + mkoff, memv + mkoff, nullptr, nullptr, mask, SS, ab);
    gemm_partial<<<gG1, tG, 0, stream>>>(ab, woc + woff, woc + woff, woc + woff,
                                         part, DD, DD, 1.0f, 1.0f, 1.0f);
    reduceK<true, false, true, false, false><<<gRd, tRow, 0, stream>>>(part, nullptr, x, nullptr, DD);

    // ---- FFN ----
    ln16<<<gRow, tRow, 0, stream>>>(x, flng + (size_t)l * DD, flnb + (size_t)l * DD, hbuf, nullptr);
    gemm_partial<<<gGf, tG, 0, stream>>>(hbuf, w1 + (size_t)l * DD * FF, w1 + (size_t)l * DD * FF,
                                         w1 + (size_t)l * DD * FF, part, DD, FF, 1.0f, 1.0f, 1.0f);
    reduceK<false, true, false, true, true><<<gRf, tRow, 0, stream>>>(part, b1 + (size_t)l * FF,
                                                                     nullptr, fbuf, FF);
    gemm_partial<<<gG1, tG, 0, stream>>>(fbuf, w2 + (size_t)l * FF * DD, w2 + (size_t)l * FF * DD,
                                         w2 + (size_t)l * FF * DD, part, FF, DD, 1.0f, 1.0f, 1.0f);
    reduceK<true, false, true, false, true><<<gRd, tRow, 0, stream>>>(part, b2 + (size_t)l * DD,
                                                                     x, nullptr, DD);
  }

  ln16<<<gRow, tRow, 0, stream>>>(x, fg, fb, nullptr, (float*)d_out);
}
